// RBFConv2d_34565896799011
// MI455X (gfx1250) — compile-verified, matrix operand order checked by reading, not run on hardware
//
#include <hip/hip_runtime.h>

typedef __attribute__((ext_vector_type(16))) _Float16 v16h;
typedef __attribute__((ext_vector_type(8)))  float    v8f;

#define B_   32
#define H_   128
#define W_   128
#define OC   512
#define KS   7
#define KK   49     // 7*7
#define OH   122
#define OW   122
#define KPAD 64     // K padded for two 16x16x32 f16 WMMA steps

// ---------------- workspace layout (bytes) ----------------
// gray : B_*H_*W_ * 4            = 2,097,152   @ 0
// x2   : B_*OH*OW * 4            = 1,905,152   @ 2,097,152
// wA   : OC*KPAD * 2 (f16 swz)   =    65,536   @ 4,002,304
// w2   : OC * 4                  =     2,048   @ 4,067,840
// misc : inv2std2 (1 float)      =       256   @ 4,069,888
#define WS_GRAY 0
#define WS_X2   2097152
#define WS_WA   4002304
#define WS_W2   4067840
#define WS_MISC 4069888

// ---- grayscale: gray[b,h,w] = .2989*R + .587*G + .114*B ----
__global__ void k_gray(const float* __restrict__ x, float* __restrict__ gray) {
    int idx = blockIdx.x * 256 + threadIdx.x;
    if (idx >= B_ * H_ * W_) return;
    int hw = idx % (H_ * W_);
    int b  = idx / (H_ * W_);
    const float* xb = x + (size_t)b * 3 * H_ * W_;
    gray[idx] = 0.2989f * xb[hw] + 0.587f * xb[H_ * W_ + hw] + 0.114f * xb[2 * H_ * W_ + hw];
}

// ---- x2[b,h,w] = sum over 7x7 window of gray^2 ----
__global__ void k_x2(const float* __restrict__ gray, float* __restrict__ x2) {
    int idx = blockIdx.x * 256 + threadIdx.x;
    if (idx >= B_ * OH * OW) return;
    int w = idx % OW;
    int h = (idx / OW) % OH;
    int b = idx / (OH * OW);
    const float* g = gray + (size_t)b * H_ * W_;
    float s = 0.f;
#pragma unroll
    for (int r = 0; r < KS; ++r)
#pragma unroll
        for (int c = 0; c < KS; ++c) {
            float v = g[(h + r) * W_ + (w + c)];
            s = fmaf(v, v, s);
        }
    x2[idx] = s;
}

// ---- prep: w2 per filter, inv2std2 scalar, f16 weights pre-swizzled into
//      WMMA A-fragment order.
// A-fragment (16-bit A 16x32, wave32): lane l holds row m = l%16; half j maps to
// lk = base8 + (j<8 ? j : 16+j-8), base8 = (l/16)*8; global k = kt*32 + lk.
// Linear element index e = ((fg*2 + kt)*32 + lane)*16 + j  -> one v16h per lane.
__global__ void k_prep(const float* __restrict__ wt, const float* __restrict__ stdp,
                       _Float16* __restrict__ wA, float* __restrict__ w2,
                       float* __restrict__ misc) {
    int tid = threadIdx.x;
    if (tid == 0) {
        float s = stdp[0];
        misc[0] = 1.0f / (2.0f * s * s);    // div done once here, not per GEMM thread
    }
    for (int o = tid; o < OC; o += 256) {
        float s = 0.f;
        for (int k = 0; k < KK; ++k) { float v = wt[o * KK + k]; s = fmaf(v, v, s); }
        w2[o] = s;
    }
    for (int e = tid; e < OC * KPAD; e += 256) {
        int j    = e & 15;
        int lane = (e >> 4) & 31;
        int kt   = (e >> 9) & 1;
        int fg   = e >> 10;                 // 16-filter group 0..31
        int m    = fg * 16 + (lane & 15);
        int base8 = (lane >> 4) * 8;
        int lk    = (j < 8) ? (base8 + j) : (base8 + 16 + (j - 8));
        int k     = kt * 32 + lk;
        wA[e] = (k < KK) ? (_Float16)wt[m * KK + k] : (_Float16)0.f;
    }
}

// ---- main: WMMA GEMM (M=512 filters, K=64 padded patch, N=64 positions/block) ----
__global__ __launch_bounds__(256) void k_gemm(const float* __restrict__ gray,
                                              const float* __restrict__ x2,
                                              const _Float16* __restrict__ wA,
                                              const float* __restrict__ w2,
                                              const float* __restrict__ misc,
                                              float* __restrict__ out) {
    __shared__ float    gtile[KS][72];      // gray strip: rows h..h+6, cols wbase..wbase+69
    __shared__ _Float16 bsh[8 * 512];       // 8 B-fragments (4 nt x 2 kt), swizzled

    const int b     = blockIdx.z;
    const int h     = blockIdx.y;
    const int wbase = blockIdx.x * 64;
    const int tid   = threadIdx.x;

    // stage gray strip (7 x 70 used, zero-pad past image edge)
    for (int i = tid; i < KS * 70; i += 256) {
        int r = i / 70, c = i % 70;
        int gw = wbase + c;
        gtile[r][c] = (gw < W_) ? gray[((size_t)b * H_ + (h + r)) * W_ + gw] : 0.f;
    }
    __syncthreads();

    // build shared B fragments: element e -> (nt, kt, lane, j) -> patch index p = r*7+c
    for (int e = tid; e < 8 * 512; e += 256) {
        int j    = e & 15;
        int lane = (e >> 4) & 31;
        int kt   = (e >> 9) & 1;
        int nt   = e >> 10;                 // N-subtile 0..3
        int n    = lane & 15;
        int base8 = (lane >> 4) * 8;
        int lk    = (j < 8) ? (base8 + j) : (base8 + 16 + (j - 8));
        int p     = kt * 32 + lk;
        _Float16 v = (_Float16)0.f;
        if (p < KK) {
            int r = p / KS, c = p % KS;
            v = (_Float16)gtile[r][nt * 16 + n + c];
        }
        bsh[e] = v;
    }
    __syncthreads();

    const int wave = tid >> 5;
    const int lane = tid & 31;

    // each wave loads all 8 shared B fragments (identical across waves) via ds_load_b128
    v16h bf[8];
#pragma unroll
    for (int f = 0; f < 8; ++f)
        bf[f] = *reinterpret_cast<const v16h*>(&bsh[f * 512 + lane * 16]);

    const float inv  = misc[0];             // uniform scalar load, no per-thread division
    const int   n    = lane & 15;
    const int   mAdd = (lane >> 4) * 8;     // C layout: lanes 16-31 hold M = v+8

    float x2v[4];
    int   wpos[4];
#pragma unroll
    for (int nt = 0; nt < 4; ++nt) {
        wpos[nt] = wbase + nt * 16 + n;
        x2v[nt]  = (wpos[nt] < OW) ? x2[((size_t)b * OH + h) * OW + wpos[nt]] : 0.f;
    }

    const v16h* wAv = reinterpret_cast<const v16h*>(wA);
#pragma unroll
    for (int mg = 0; mg < 4; ++mg) {
        const int fg = mg * 8 + wave;       // 16-filter group handled by this wave
        v16h a0 = wAv[(fg * 2 + 0) * 32 + lane];   // K = 0..31   (global_load_b128 x2)
        v16h a1 = wAv[(fg * 2 + 1) * 32 + lane];   // K = 32..63

        const int m0 = fg * 16;
        float w2v[8];
#pragma unroll
        for (int v = 0; v < 8; ++v) w2v[v] = w2[m0 + mAdd + v];

#pragma unroll
        for (int nt = 0; nt < 4; ++nt) {
            v8f c = {};
            c = __builtin_amdgcn_wmma_f32_16x16x32_f16(false, a0, false, bf[nt * 2 + 0],
                                                       (short)0, c, false, false);
            c = __builtin_amdgcn_wmma_f32_16x16x32_f16(false, a1, false, bf[nt * 2 + 1],
                                                       (short)0, c, false, false);
            if (wpos[nt] < OW) {
#pragma unroll
                for (int v = 0; v < 8; ++v) {
                    int   o  = m0 + mAdd + v;
                    float sq = x2v[nt] + w2v[v] - 2.0f * c[v];
                    sq = fmaxf(sq, 0.f);
                    out[(((size_t)b * OC + o) * OH + h) * OW + wpos[nt]] = __expf(-sq * inv);
                }
            }
        }
    }
}

extern "C" void kernel_launch(void* const* d_in, const int* in_sizes, int n_in,
                              void* d_out, int out_size, void* d_ws, size_t ws_size,
                              hipStream_t stream) {
    const float* x    = (const float*)d_in[0];   // [32,3,128,128]
    const float* wt   = (const float*)d_in[1];   // [512,1,7,7]
    const float* stdp = (const float*)d_in[2];   // scalar

    char* ws = (char*)d_ws;
    float*    gray = (float*)(ws + WS_GRAY);
    float*    x2   = (float*)(ws + WS_X2);
    _Float16* wA   = (_Float16*)(ws + WS_WA);
    float*    w2   = (float*)(ws + WS_W2);
    float*    misc = (float*)(ws + WS_MISC);
    float*    out  = (float*)d_out;

    k_prep<<<1, 256, 0, stream>>>(wt, stdp, wA, w2, misc);
    k_gray<<<(B_ * H_ * W_ + 255) / 256, 256, 0, stream>>>(x, gray);
    k_x2<<<(B_ * OH * OW + 255) / 256, 256, 0, stream>>>(gray, x2);
    k_gemm<<<dim3(2, OH, B_), 256, 0, stream>>>(gray, x2, wA, w2, misc, out);
}
